// GraphTransformer2DModel_7413113553210
// MI455X (gfx1250) — matrix operand order repaired and checked by reading, not compiled, not run on hardware
//
#include <hip/hip_runtime.h>
#include <hip/hip_bf16.h>
#include <math.h>

// ---------------------------------------------------------------------------
// GraphTransformer2D forward on gfx1250 (CDNA5, wave32).
// Dense GEMMs: v_wmma_f32_16x16x32_bf16 on pre-converted bf16 operand buffers
// (activations converted once per GEMM, weights transposed/converted once).
// Inner loop: all loads (A: 2x b128, B: 64x u16) issued up front, then 4 WMMAs -> partial
// loadcnt waits give load/compute overlap.
// Irregular parts (edge segment softmax, tiny MHA, layernorm) on VALU/atomics.
// ---------------------------------------------------------------------------

typedef __attribute__((ext_vector_type(16))) __bf16 v16bf;
typedef __attribute__((ext_vector_type(8)))  float  v8f;

#define NTOT  32768   // B*N nodes
#define DM    256     // model dim
#define NH    8       // heads
#define HDIM  32      // head dim
#define NE    524288  // edges
#define BSZ   256
#define NNODE 128
#define SS    8
#define FE    16

__device__ __forceinline__ unsigned short bfbits(float f) {
  union { float f; unsigned u; } in; in.f = f;
  unsigned u = in.u;
  return (unsigned short)((u + 0x7fffu + ((u >> 16) & 1u)) >> 16);  // RNE
}

// ---------------------------------------------------------------------------
// Aligned bf16 WMMA GEMM: C[M,N] = act(A[M,K] @ B[K,N] + bias)
// A, B are bf16 (row-major); C/bias f32. M%16==0, N%64==0, K%32==0.
// One wave -> 16x64 C tile (4 accumulators share one A fragment per K step).
// Fragment layouts per CDNA5 ISA 7.12.2:
//   A 16x32 bf16 : lane m=l&15, hl=l>>4; elem j -> K=8*hl+j, elem j+8 -> K=16+8*hl+j
//   B 32x16 bf16 : lane col n=l&15; elem s -> same K as A elem s
//   C 16x16 f32  : lane col n=l&15, VGPR v -> row v+8*(l>>4)
// ---------------------------------------------------------------------------
__global__ void __launch_bounds__(128)
gemm_bf_n64(const unsigned short* __restrict__ A, int lda,
            const unsigned short* __restrict__ B, int ldb,
            const float* __restrict__ bias,
            float* __restrict__ C, int ldc,
            int M, int N, int K, int act) {
  int wave = blockIdx.x * (blockDim.x >> 5) + (threadIdx.x >> 5);
  int lane = threadIdx.x & 31;
  int tilesN = N >> 6;                       // 64-wide tiles
  if (wave >= (M >> 4) * tilesN) return;     // wave-uniform: EXEC stays full
  int m0 = (wave / tilesN) << 4;
  int n0 = (wave % tilesN) << 6;

  int mrow = lane & 15;
  int hl   = lane >> 4;
  const unsigned short* Arow = A + (size_t)(m0 + mrow) * lda + 8 * hl;
  const unsigned short* Bbase = B + n0;

  v8f acc[4];
  #pragma unroll
  for (int t = 0; t < 4; ++t) acc[t] = (v8f){0.f,0.f,0.f,0.f,0.f,0.f,0.f,0.f};

  for (int kk = 0; kk < K; kk += 32) {
    // issue ALL loads first (A: 2x b128, B: 64x u16) so WMMAs can overlap
    // with in-flight loads via partial loadcnt waits.
    union { v16bf v; uint4 u[2]; } a, b0, b1, b2, b3;
    a.u[0] = *(const uint4*)(Arow + kk);        // K = kk + 8*hl + [0..7]
    a.u[1] = *(const uint4*)(Arow + kk + 16);   // K = kk + 16 + 8*hl + [0..7]

    const unsigned short* Bcol = Bbase + mrow + (size_t)(kk + 8 * hl) * ldb;
    #pragma unroll
    for (int p = 0; p < 8; ++p) {
      int s0 = 2 * p, s1 = 2 * p + 1;
      size_t o0 = (size_t)(s0 < 8 ? s0 : s0 + 8) * ldb;
      size_t o1 = (size_t)(s1 < 8 ? s1 : s1 + 8) * ldb;
      unsigned w0 = (unsigned)Bcol[o0 + 0]  | ((unsigned)Bcol[o1 + 0]  << 16);
      unsigned w1 = (unsigned)Bcol[o0 + 16] | ((unsigned)Bcol[o1 + 16] << 16);
      unsigned w2 = (unsigned)Bcol[o0 + 32] | ((unsigned)Bcol[o1 + 32] << 16);
      unsigned w3 = (unsigned)Bcol[o0 + 48] | ((unsigned)Bcol[o1 + 48] << 16);
      ((unsigned*)&b0.u[p >> 2])[p & 3] = w0;
      ((unsigned*)&b1.u[p >> 2])[p & 3] = w1;
      ((unsigned*)&b2.u[p >> 2])[p & 3] = w2;
      ((unsigned*)&b3.u[p >> 2])[p & 3] = w3;
    }

    acc[0] = __builtin_amdgcn_wmma_f32_16x16x32_bf16(false, a.v, false, b0.v,
                                                     (short)0, acc[0], false, false);
    acc[1] = __builtin_amdgcn_wmma_f32_16x16x32_bf16(false, a.v, false, b1.v,
                                                     (short)0, acc[1], false, false);
    acc[2] = __builtin_amdgcn_wmma_f32_16x16x32_bf16(false, a.v, false, b2.v,
                                                     (short)0, acc[2], false, false);
    acc[3] = __builtin_amdgcn_wmma_f32_16x16x32_bf16(false, a.v, false, b3.v,
                                                     (short)0, acc[3], false, false);
  }

  // store 4 16x16 tiles
  int nc = lane & 15;
  #pragma unroll
  for (int t = 0; t < 4; ++t) {
    int col = n0 + 16 * t + nc;
    float bv = bias ? bias[col] : 0.f;
    #pragma unroll
    for (int vv = 0; vv < 8; ++vv) {
      int row = m0 + vv + 8 * hl;
      float o = acc[t][vv] + bv;
      if (act == 1) o = fmaxf(o, 0.f);
      C[(size_t)row * ldc + col] = o;
    }
  }
}

// ----------------------- operand conversion kernels -------------------------

// flat f32 -> bf16, 4 elements/thread (n % 4 == 0)
__global__ void cvt_bf(const float* __restrict__ src, unsigned short* __restrict__ dst,
                       long n) {
  long i = ((long)blockIdx.x * blockDim.x + threadIdx.x) * 4;
  if (i >= n) return;
  float4 f = *(const float4*)(src + i);
  uint2 o;
  o.x = (unsigned)bfbits(f.x) | ((unsigned)bfbits(f.y) << 16);
  o.y = (unsigned)bfbits(f.z) | ((unsigned)bfbits(f.w) << 16);
  *(uint2*)(dst + i) = o;
}

// W [N,K] row-major -> WT [K,N] bf16 (for x @ W.T)
__global__ void transpose_cvt_bf(const float* __restrict__ W, unsigned short* __restrict__ WT,
                                 int N, int K) {
  int idx = blockIdx.x * blockDim.x + threadIdx.x;
  if (idx >= N * K) return;
  int k = idx / N, n = idx % N;
  WT[idx] = bfbits(W[(size_t)n * K + k]);
}

// per-head weight [H, D, HD] -> [D, H*HD] bf16
__global__ void repack_hde_bf(const float* __restrict__ W, unsigned short* __restrict__ out) {
  int idx = blockIdx.x * blockDim.x + threadIdx.x;
  if (idx >= DM * DM) return;
  int d = idx >> 8;
  int n = idx & 255;
  int h = n >> 5;
  int e = n & 31;
  out[idx] = bfbits(W[(size_t)h * DM * HDIM + (size_t)d * HDIM + e]);
}

// zero-pad x [NTOT,16] -> bf16 [NTOT,32]
__global__ void pad_x16_bf(const float* __restrict__ x, unsigned short* __restrict__ xp) {
  int r = blockIdx.x, t = threadIdx.x;    // 32 threads
  xp[(size_t)r * 32 + t] = (t < 16) ? bfbits(x[(size_t)r * 16 + t]) : (unsigned short)0;
}

// zero-pad in_w [16,256] -> bf16 [32,256]
__global__ void pad_w16_bf(const float* __restrict__ w, unsigned short* __restrict__ wp) {
  int idx = blockIdx.x * blockDim.x + threadIdx.x;   // 32*256
  if (idx >= 32 * 256) return;
  wp[idx] = (idx < 16 * 256) ? bfbits(w[idx]) : (unsigned short)0;
}

// --------------------------- small helper kernels ---------------------------

__global__ void zero_kernel(float* __restrict__ p, long n) {
  long i = (long)blockIdx.x * blockDim.x + threadIdx.x;
  if (i < n) p[i] = 0.f;
}

__global__ void vec_add(float* __restrict__ x, const float* __restrict__ y, long n) {
  long i = (long)blockIdx.x * blockDim.x + threadIdx.x;
  if (i < n) x[i] += y[i];
}

// x[r, d] += pos_enc[r % 128, d]   (grid = NTOT blocks of 256)
__global__ void add_pos(float* __restrict__ x, const float* __restrict__ pos) {
  int r = blockIdx.x, d = threadIdx.x;
  x[(size_t)r * DM + d] += pos[((r & (NNODE - 1)) << 8) + d];
}

// O[r] = LN(X[r] + R[r]) * g + b      (one 256-thread block per row; O may alias R)
__global__ void ln_res(const float* __restrict__ X, const float* __restrict__ R,
                       const float* __restrict__ g, const float* __restrict__ b,
                       float* __restrict__ O) {
  int r = blockIdx.x, d = threadIdx.x;
  __shared__ float red[DM];
  float v = X[(size_t)r * DM + d] + R[(size_t)r * DM + d];
  red[d] = v; __syncthreads();
  for (int s = 128; s > 0; s >>= 1) { if (d < s) red[d] += red[d + s]; __syncthreads(); }
  float mu = red[0] * (1.f / DM); __syncthreads();
  float dv = v - mu;
  red[d] = dv * dv; __syncthreads();
  for (int s = 128; s > 0; s >>= 1) { if (d < s) red[d] += red[d + s]; __syncthreads(); }
  float var = red[0] * (1.f / DM);
  O[(size_t)r * DM + d] = dv * rsqrtf(var + 1e-5f) * g[d] + b[d];
}

// ------------------------ edge (graph attention) path -----------------------
// QKV buffer layout: row stride 768, cols [0,256)=Q, [256,512)=K, [512,768)=V.

__global__ void edge_scores(const float* __restrict__ QKV, const int* __restrict__ src,
                            const int* __restrict__ dst, float* __restrict__ ES) {
  long t = (long)blockIdx.x * blockDim.x + threadIdx.x;
  if (t >= (long)NE * NH) return;
  int e = (int)(t >> 3), h = (int)(t & 7);
  const float* q = QKV + (size_t)dst[e] * 768 + h * HDIM;
  const float* k = QKV + (size_t)src[e] * 768 + 256 + h * HDIM;
  float acc = 0.f;
  #pragma unroll 8
  for (int i = 0; i < HDIM; ++i) acc += q[i] * k[i];
  ES[t] = acc * 0.17677669529663687f;   // 1/sqrt(32)
}

__global__ void seg_init(float* __restrict__ m, float* __restrict__ d, int n) {
  int i = blockIdx.x * blockDim.x + threadIdx.x;
  if (i < n) { m[i] = -__builtin_inff(); d[i] = 0.f; }
}

__global__ void edge_max(const float* __restrict__ ES, const int* __restrict__ dst,
                         float* __restrict__ m) {
  long t = (long)blockIdx.x * blockDim.x + threadIdx.x;
  if (t >= (long)NE * NH) return;
  int e = (int)(t >> 3), h = (int)(t & 7);
  float v = ES[t];
  float* addr = m + (size_t)dst[e] * NH + h;
  if (v >= 0.f) atomicMax((int*)addr, __float_as_int(v));
  else          atomicMin((unsigned int*)addr, __float_as_uint(v));
}

__global__ void edge_exp(float* __restrict__ ES, const int* __restrict__ dst,
                         const float* __restrict__ m, float* __restrict__ dsum) {
  long t = (long)blockIdx.x * blockDim.x + threadIdx.x;
  if (t >= (long)NE * NH) return;
  int e = (int)(t >> 3), h = (int)(t & 7);
  float ex = __expf(ES[t] - m[(size_t)dst[e] * NH + h]);
  ES[t] = ex;
  atomicAdd(dsum + (size_t)dst[e] * NH + h, ex);
}

__global__ void edge_msg(const float* __restrict__ ES, const int* __restrict__ src,
                         const int* __restrict__ dst, const float* __restrict__ dsum,
                         const float* __restrict__ QKV, float* __restrict__ msg) {
  long t = (long)blockIdx.x * blockDim.x + threadIdx.x;
  if (t >= (long)NE * NH) return;
  int e = (int)(t >> 3), h = (int)(t & 7);
  int dn = dst[e];
  float alpha = ES[t] / (dsum[(size_t)dn * NH + h] + 1e-16f);
  const float* v = QKV + (size_t)src[e] * 768 + 512 + h * HDIM;
  float* o = msg + (size_t)dn * DM + h * HDIM;
  #pragma unroll 8
  for (int i = 0; i < HDIM; ++i) atomicAdd(o + i, alpha * v[i]);
}

// --------------------- tiny-sequence MHA (L = 8 or 16) ----------------------
// One wave per (sequence-group, head). QKV rows live at
// row(s) = (g/sub)*128 + (g%sub)*substride + s*stride  (absorbs the reshapes).
__global__ void mha_attn(const float* __restrict__ QKV, float* __restrict__ O,
                         int G, int L, int sub, int substride, int stride) {
  int wave = blockIdx.x * (blockDim.x >> 5) + (threadIdx.x >> 5);
  int lane = threadIdx.x & 31;
  if (wave >= G * NH) return;                    // wave-uniform
  int g = wave >> 3, h = wave & 7;
  int base = (g / sub) * NNODE + (g % sub) * substride;
  const float scale = 0.17677669529663687f;
  for (int i = 0; i < L; ++i) {
    int ri = base + i * stride;
    const float* qi = QKV + (size_t)ri * 768 + h * HDIM;
    float s = -__builtin_inff();
    if (lane < L) {
      const float* kj = QKV + (size_t)(base + lane * stride) * 768 + 256 + h * HDIM;
      float acc = 0.f;
      #pragma unroll 8
      for (int d = 0; d < HDIM; ++d) acc += qi[d] * kj[d];
      s = acc * scale;
    }
    float mx = s;
    for (int o = 16; o > 0; o >>= 1) mx = fmaxf(mx, __shfl_xor(mx, o, 32));
    float e = (lane < L) ? __expf(s - mx) : 0.f;
    float sm = e;
    for (int o = 16; o > 0; o >>= 1) sm += __shfl_xor(sm, o, 32);
    float alpha = e / sm;
    float od = 0.f;                              // lane = head-dim index (0..31)
    for (int j = 0; j < L; ++j) {
      float aj = __shfl(alpha, j, 32);
      const float* vj = QKV + (size_t)(base + j * stride) * 768 + 512 + h * HDIM;
      od += aj * vj[lane];
    }
    O[(size_t)ri * DM + h * HDIM + lane] = od;
  }
}

// final head: out[r] = sigmoid(dot(X[r,:K], w) + b)
__global__ void head_out(const float* __restrict__ X, const float* __restrict__ w,
                         const float* __restrict__ b, float* __restrict__ out,
                         int rows, int K) {
  int r = blockIdx.x * blockDim.x + threadIdx.x;
  if (r >= rows) return;
  float acc = b[0];
  for (int k = 0; k < K; ++k) acc += X[(size_t)r * K + k] * w[k];
  out[r] = 1.f / (1.f + __expf(-acc));
}

// ---------------------------------------------------------------------------

extern "C" void kernel_launch(void* const* d_in, const int* in_sizes, int n_in,
                              void* d_out, int out_size, void* d_ws, size_t ws_size,
                              hipStream_t stream) {
  (void)in_sizes; (void)n_in; (void)out_size; (void)ws_size;

  const float* x    = (const float*)d_in[0];
  const int*   eidx = (const int*)d_in[1];
  const float* pos  = (const float*)d_in[3];
  const float* in_w = (const float*)d_in[4];
  const float* in_b = (const float*)d_in[5];
  const int* src = eidx;
  const int* dst = eidx + NE;

  // workspace carve-up (floats)
  float* ws  = (float*)d_ws;
  constexpr size_t NT = NTOT;
  float* XA  = ws;                          // activations (ping) [NT,256]
  float* XB  = XA  + NT * 256;              // activations (pong)
  float* T0  = XB  + NT * 256;              // QKV scratch [NT,768]
  float* TSK = T0  + NT * 768;              // graph skip  [NT,256]
  float* T1  = TSK + NT * 256;              // msg / attn-out [NT,256]
  float* ES  = T1  + NT * 256;              // edge scores [E,H]
  float* NMm = ES  + (size_t)NE * NH;       // segment max [NT,H]
  float* NMd = NMm + NT * NH;               // segment sum [NT,H]
  float* A1  = NMd + NT * NH;               // [256,1024]
  float* A2  = A1  + 256 * 1024;            // [256,512]
  float* A3  = A2  + 256 * 512;             // [256,256]
  float* Hb1 = A3  + 256 * 256;             // [256,128]
  float* Hb2 = Hb1 + 256 * 128;             // [256,64]
  unsigned short* ABF = (unsigned short*)(Hb2 + 256 * 64);  // bf16 A buf [NT,512] max
  unsigned short* WBF = ABF + NT * 512;                     // bf16 W buf [32768,1024] max

  auto gemm = [&](const unsigned short* A, int lda, const unsigned short* B, int ldb,
                  const float* bias, float* C, int ldc, int M, int N, int K, int act) {
    int waves = (M >> 4) * (N >> 6);
    int blocks = (waves + 3) / 4;     // 4 waves / 128-thread block
    gemm_bf_n64<<<blocks, 128, 0, stream>>>(A, lda, B, ldb, bias, C, ldc, M, N, K, act);
  };
  auto cvt = [&](const float* s, unsigned short* d, long n) {
    cvt_bf<<<(int)((n / 4 + 255) / 256), 256, 0, stream>>>(s, d, n);
  };
  auto tcvt = [&](const float* w, unsigned short* d, int N, int K) {
    transpose_cvt_bf<<<(N * K + 255) / 256, 256, 0, stream>>>(w, d, N, K);
  };

  const long EH = (long)NE * NH;
  const int  eb = (int)((EH + 255) / 256);
  const int  nb = (int)(((long)NT * 256 + 255) / 256);

  // ---- input projection (zero-padded to K=32) + positional encoding ----
  pad_x16_bf<<<NTOT, 32, 0, stream>>>(x, ABF);
  pad_w16_bf<<<32, 256, 0, stream>>>(in_w, WBF);
  gemm(ABF, 32, WBF, 256, in_b, XA, 256, NTOT, 256, 32, 0);
  add_pos<<<NTOT, 256, 0, stream>>>(XA, pos);

  // ---- 2 graph (TransformerConv) layers ----
  for (int l = 0; l < 2; ++l) {
    int gi = 6 + 12 * l;
    const float* Wq = (const float*)d_in[gi + 0];
    const float* bq = (const float*)d_in[gi + 1];
    const float* Wk = (const float*)d_in[gi + 2];
    const float* bk = (const float*)d_in[gi + 3];
    const float* Wv = (const float*)d_in[gi + 4];
    const float* bv = (const float*)d_in[gi + 5];
    const float* Wsk= (const float*)d_in[gi + 6];
    const float* bs = (const float*)d_in[gi + 7];
    const float* Wo = (const float*)d_in[gi + 8];
    const float* bo = (const float*)d_in[gi + 9];
    const float* gg = (const float*)d_in[gi + 10];
    const float* gb = (const float*)d_in[gi + 11];

    cvt(XA, ABF, (long)NT * 256);                                        // x -> bf16 once
    repack_hde_bf<<<256, 256, 0, stream>>>(Wq, WBF);
    gemm(ABF, 256, WBF, 256, bq, T0 + 0,   768, NTOT, 256, 256, 0);      // Q
    repack_hde_bf<<<256, 256, 0, stream>>>(Wk, WBF);
    gemm(ABF, 256, WBF, 256, bk, T0 + 256, 768, NTOT, 256, 256, 0);      // K
    repack_hde_bf<<<256, 256, 0, stream>>>(Wv, WBF);
    gemm(ABF, 256, WBF, 256, bv, T0 + 512, 768, NTOT, 256, 256, 0);      // V
    repack_hde_bf<<<256, 256, 0, stream>>>(Wsk, WBF);
    gemm(ABF, 256, WBF, 256, bs, TSK, 256, NTOT, 256, 256, 0);           // skip

    edge_scores<<<eb, 256, 0, stream>>>(T0, src, dst, ES);
    seg_init<<<(NTOT * NH + 255) / 256, 256, 0, stream>>>(NMm, NMd, NTOT * NH);
    edge_max<<<eb, 256, 0, stream>>>(ES, dst, NMm);
    edge_exp<<<eb, 256, 0, stream>>>(ES, dst, NMm, NMd);
    zero_kernel<<<nb, 256, 0, stream>>>(T1, (long)NT * 256);
    edge_msg<<<eb, 256, 0, stream>>>(ES, src, dst, NMd, T0, T1);
    vec_add<<<nb, 256, 0, stream>>>(T1, TSK, (long)NT * 256);            // msg + skip

    cvt(T1, ABF, (long)NT * 256);
    cvt(Wo, WBF, 256 * 256);                                             // [K,N] direct
    gemm(ABF, 256, WBF, 256, bo, XB, 256, NTOT, 256, 256, 0);            // out proj
    ln_res<<<NTOT, 256, 0, stream>>>(XB, XA, gg, gb, XA);                // LN(out + x)
  }

  // ---- 2 spatial-temporal layers ----
  for (int l = 0; l < 2; ++l) {
    int si = 30 + 18 * l;
    const float* st_wi = (const float*)d_in[si + 0];
    const float* st_bi = (const float*)d_in[si + 1];
    const float* st_wo = (const float*)d_in[si + 2];
    const float* st_bo = (const float*)d_in[si + 3];
    const float* ft_wi = (const float*)d_in[si + 4];
    const float* ft_bi = (const float*)d_in[si + 5];
    const float* ft_wo = (const float*)d_in[si + 6];
    const float* ft_bo = (const float*)d_in[si + 7];
    const float* w1    = (const float*)d_in[si + 8];
    const float* b1    = (const float*)d_in[si + 9];
    const float* w2    = (const float*)d_in[si + 10];
    const float* b2    = (const float*)d_in[si + 11];
    const float* g1    = (const float*)d_in[si + 12];
    const float* b1n   = (const float*)d_in[si + 13];
    const float* g2    = (const float*)d_in[si + 14];
    const float* b2n   = (const float*)d_in[si + 15];
    const float* g3    = (const float*)d_in[si + 16];
    const float* b3n   = (const float*)d_in[si + 17];

    // temporal attention over S=8 (groups = Bsz*Fe = 4096)
    cvt(XA, ABF, (long)NT * 256);
    tcvt(st_wi, WBF, 768, 256);                                          // wi.T -> [256,768]
    gemm(ABF, 256, WBF, 768, st_bi, T0, 768, NTOT, 768, 256, 0);
    { int waves = (BSZ * FE) * NH;
      mha_attn<<<(waves + 3) / 4, 128, 0, stream>>>(T0, T1, BSZ * FE, SS, FE, 1, FE); }
    cvt(T1, ABF, (long)NT * 256);
    tcvt(st_wo, WBF, 256, 256);                                          // wo.T
    gemm(ABF, 256, WBF, 256, st_bo, XB, 256, NTOT, 256, 256, 0);
    ln_res<<<NTOT, 256, 0, stream>>>(XB, XA, g1, b1n, XA);

    // feature attention over Fe=16 (groups = Bsz*S = 2048)
    cvt(XA, ABF, (long)NT * 256);
    tcvt(ft_wi, WBF, 768, 256);
    gemm(ABF, 256, WBF, 768, ft_bi, T0, 768, NTOT, 768, 256, 0);
    { int waves = (BSZ * SS) * NH;
      mha_attn<<<(waves + 3) / 4, 128, 0, stream>>>(T0, T1, BSZ * SS, FE, SS, FE, 1); }
    cvt(T1, ABF, (long)NT * 256);
    tcvt(ft_wo, WBF, 256, 256);
    gemm(ABF, 256, WBF, 256, ft_bo, XB, 256, NTOT, 256, 256, 0);
    ln_res<<<NTOT, 256, 0, stream>>>(XB, XA, g2, b2n, XA);

    // FFN 256 -> 512 -> 256
    cvt(XA, ABF, (long)NT * 256);
    cvt(w1, WBF, 256 * 512);
    gemm(ABF, 256, WBF, 512, b1, T0, 512, NTOT, 512, 256, 1);
    cvt(T0, ABF, (long)NT * 512);
    cvt(w2, WBF, 512 * 256);
    gemm(ABF, 512, WBF, 256, b2, XB, 256, NTOT, 256, 512, 0);
    ln_res<<<NTOT, 256, 0, stream>>>(XB, XA, g3, b3n, XA);
  }

  // ---- aggregator MLP (flat view [256, 32768]) + head ----
  const float* aw1 = (const float*)d_in[66]; const float* ab1 = (const float*)d_in[67];
  const float* aw2 = (const float*)d_in[68]; const float* ab2 = (const float*)d_in[69];
  const float* aw3 = (const float*)d_in[70]; const float* ab3 = (const float*)d_in[71];
  const float* hw1 = (const float*)d_in[72]; const float* hb1 = (const float*)d_in[73];
  const float* hw2 = (const float*)d_in[74]; const float* hb2 = (const float*)d_in[75];
  const float* hw3 = (const float*)d_in[76]; const float* hb3 = (const float*)d_in[77];

  cvt(XA, ABF, (long)NT * 256);                   // flat [256, 32768]
  cvt(aw1, WBF, (long)32768 * 1024);
  gemm(ABF, 32768, WBF, 1024, ab1, A1, 1024, BSZ, 1024, 32768, 1);
  cvt(A1, ABF, 256 * 1024);
  cvt(aw2, WBF, 1024 * 512);
  gemm(ABF, 1024, WBF, 512, ab2, A2, 512, BSZ, 512, 1024, 1);
  cvt(A2, ABF, 256 * 512);
  cvt(aw3, WBF, 512 * 256);
  gemm(ABF, 512, WBF, 256, ab3, A3, 256, BSZ, 256, 512, 1);
  cvt(A3, ABF, 256 * 256);
  cvt(hw1, WBF, 256 * 128);
  gemm(ABF, 256, WBF, 128, hb1, Hb1, 128, BSZ, 128, 256, 1);
  cvt(Hb1, ABF, 256 * 128);
  cvt(hw2, WBF, 128 * 64);
  gemm(ABF, 128, WBF, 64, hb2, Hb2, 64, BSZ, 64, 128, 1);
  head_out<<<1, 256, 0, stream>>>(Hb2, hw3, hb3, (float*)d_out, BSZ, 64);
}
